// MyMSA_21251498181323
// MI455X (gfx1250) — matrix operand order repaired and checked by reading, not compiled, not run on hardware
//
#include <hip/hip_runtime.h>

typedef __attribute__((ext_vector_type(16))) _Float16 v16h;
typedef __attribute__((ext_vector_type(8)))  _Float16 v8h;
typedef __attribute__((ext_vector_type(8)))  float    v8f;
typedef __attribute__((ext_vector_type(4)))  int      v4i;

static constexpr int cB = 2, cS = 4096, cD = 256, cH = 4, cDH = 64;
static constexpr int cNT = cS / 16;            // 16-row s tiles per (b,h)

// ---- optional gfx1250 async global->LDS path (ASYNCcnt) --------------------
#if defined(__has_builtin)
#  if __has_builtin(__builtin_amdgcn_global_load_async_to_lds_b128) && \
      __has_builtin(__builtin_amdgcn_s_wait_asynccnt)
#    define USE_ASYNC_LDS 1
#  endif
#endif

typedef __attribute__((address_space(1))) v4i gv4i;   // global int4
typedef __attribute__((address_space(3))) v4i lv4i;   // LDS int4

__device__ __forceinline__ void cp16_g2l(const _Float16* g, _Float16* l) {
#if defined(USE_ASYNC_LDS)
    __builtin_amdgcn_global_load_async_to_lds_b128((gv4i*)g, (lv4i*)l, 0, 0);
#else
    *(v8h*)l = *(const v8h*)g;                 // global_load_b128 + ds_store_b128
#endif
}
__device__ __forceinline__ void wait_stage() {
#if defined(USE_ASYNC_LDS)
    __builtin_amdgcn_s_wait_asynccnt(0);
#endif
}

// A-matrix (16-bit, 16xK) per-lane K index for v16h element e=2*i+j (contiguous: k = 16*hh + e)
__device__ __forceinline__ int a_klocal(int i, int j, int hh) {
    return 16 * hh + 2 * i + j;
}

// ---------------------------------------------------------------------------
// Kernel 1: per-head QKV projection, fp32 -> f16, WMMA f32_16x16x32_f16.
//   Q,K stored [B,H,S,DH] row-major f16; V stored transposed [B,H,DH,S] f16
// ---------------------------------------------------------------------------
__global__ __launch_bounds__(32, 1)
void qkv_proj_kernel(const float* __restrict__ seq,
                     const float* __restrict__ qw, const float* __restrict__ qb,
                     const float* __restrict__ kw, const float* __restrict__ kb,
                     const float* __restrict__ vw, const float* __restrict__ vb,
                     _Float16* __restrict__ Q, _Float16* __restrict__ K,
                     _Float16* __restrict__ Vt) {
    const int tile = blockIdx.x;               // B*H*cNT blocks, 1 wave each
    const int b  = tile / (cH * cNT);
    const int h  = (tile / cNT) % cH;
    const int m0 = (tile % cNT) * 16;
    const int L  = threadIdx.x;
    const int nl = L & 15;
    const int hh = L >> 4;

    // A fragments: X tile rows m0..m0+15, cols d = h*64 .. h*64+63
    v16h a0, a1;
    const float* xrow = seq + ((size_t)b * cS + (m0 + nl)) * cD + h * cDH;
#pragma unroll
    for (int i = 0; i < 8; ++i)
#pragma unroll
        for (int j = 0; j < 2; ++j) {
            const int kl = a_klocal(i, j, hh);
            a0[2 * i + j] = (_Float16)xrow[kl];
            a1[2 * i + j] = (_Float16)xrow[32 + kl];
        }

    const float* Ws[3] = { qw, kw, vw };
    const float* Bs[3] = { qb, kb, vb };

#pragma unroll 1
    for (int p = 0; p < 3; ++p) {
        const float* W    = Ws[p] + (size_t)h * cDH * cDH;   // W[e][d]
        const float* bias = Bs[p] + h * cDH;
#pragma unroll 1
        for (int g = 0; g < 4; ++g) {
            const int n0 = 16 * g;
            // B fragment: Bmat[k][n] = W[n0+nl][k]; lane reads 16 contiguous floats
            const float* wrow = W + (size_t)(n0 + nl) * cDH + 16 * hh;
            v16h b0, b1;
#pragma unroll
            for (int e = 0; e < 16; ++e) {
                b0[e] = (_Float16)wrow[e];
                b1[e] = (_Float16)wrow[32 + e];
            }
            v8f c = {};
            c = __builtin_amdgcn_wmma_f32_16x16x32_f16(false, a0, false, b0, (short)0, c, false, false);
            c = __builtin_amdgcn_wmma_f32_16x16x32_f16(false, a1, false, b1, (short)0, c, false, false);
            const float bb = bias[n0 + nl];
            if (p == 2) {  // V transposed: Vt[b,h, e, s]
                _Float16* dst = Vt + (((size_t)b * cH + h) * cDH + (n0 + nl)) * cS + m0 + 8 * hh;
#pragma unroll
                for (int r = 0; r < 8; ++r) dst[r] = (_Float16)(c[r] + bb);
            } else {       // Q/K row-major: [b,h, s, e]
                _Float16* dst = (p == 0 ? Q : K) + ((size_t)b * cH + h) * cS * cDH;
#pragma unroll
                for (int r = 0; r < 8; ++r)
                    dst[(size_t)(m0 + 8 * hh + r) * cDH + n0 + nl] = (_Float16)(c[r] + bb);
            }
        }
    }
}

// ---------------------------------------------------------------------------
// Kernel 2: flash attention. 4 waves / workgroup; 64 queries per WG.
// K/V 32-kv chunks double-buffered in LDS (async global->LDS when available).
// ---------------------------------------------------------------------------
__global__ __launch_bounds__(128, 1)
void flash_attn_kernel(const _Float16* __restrict__ Q,
                       const _Float16* __restrict__ K,
                       const _Float16* __restrict__ Vt,
                       float* __restrict__ out) {
    __shared__ alignas(16) _Float16 kbuf[2][32][64];   // [buf][kv][d]     4KB/buf
    __shared__ alignas(16) _Float16 vbuf[2][64][32];   // [buf][e][kv]     4KB/buf
    __shared__ alignas(16) _Float16 pbuf[4][16][32];   // per-wave P tile

    const int nwgt = cS / 64;
    const int wgt  = blockIdx.x;                        // cB*cH*nwgt
    const int b  = wgt / (cH * nwgt);
    const int h  = (wgt / nwgt) % cH;
    const int q0 = (wgt % nwgt) * 64;
    const int t  = threadIdx.x;
    const int w  = t >> 5;                              // wave in WG
    const int L  = t & 31;
    const int nl = L & 15;
    const int hh = L >> 4;
    const int m0 = q0 + 16 * w;                         // this wave's q rows

    const _Float16* kbase = K  + ((size_t)b * cH + h) * cS * cDH;
    const _Float16* vbase = Vt + ((size_t)b * cH + h) * cDH * cS;

    // Q A-fragments (K-dim = head dim 64 -> two 16x32 fragments)
    v16h aq0, aq1;
    const _Float16* qrow = Q + (((size_t)b * cH + h) * cS + m0 + nl) * cDH;
#pragma unroll
    for (int i = 0; i < 8; ++i)
#pragma unroll
        for (int j = 0; j < 2; ++j) {
            const int kl = a_klocal(i, j, hh);
            aq0[2 * i + j] = qrow[kl];
            aq1[2 * i + j] = qrow[32 + kl];
        }

    // cooperative stage of one 32-kv chunk (K: 32x64 f16, Vt: 64x32 f16)
    auto stage = [&](int bufi, int kv0) {
#pragma unroll
        for (int it = 0; it < 2; ++it) {                // K: 256 x 16B
            const int idx = it * 128 + t;
            cp16_g2l(kbase + (size_t)(kv0 + (idx >> 3)) * cDH + (idx & 7) * 8,
                     &kbuf[bufi][idx >> 3][(idx & 7) * 8]);
        }
#pragma unroll
        for (int it = 0; it < 2; ++it) {                // V: 256 x 16B
            const int idx = it * 128 + t;
            cp16_g2l(vbase + (size_t)(idx >> 2) * cS + kv0 + (idx & 3) * 8,
                     &vbuf[bufi][idx >> 2][(idx & 3) * 8]);
        }
    };

    stage(0, 0);
    wait_stage();
    __syncthreads();

    v8f o[4] = { v8f{}, v8f{}, v8f{}, v8f{} };
    float mrow[8], lrow[8];
#pragma unroll
    for (int r = 0; r < 8; ++r) { mrow[r] = -1e30f; lrow[r] = 0.f; }
    const float sc = 0.125f;                            // 1/sqrt(64)

#pragma unroll 1
    for (int kv0 = 0; kv0 < cS; kv0 += 32) {
        const int buf = (kv0 >> 5) & 1;
        if (kv0 + 32 < cS) stage(buf ^ 1, kv0 + 32);    // prefetch next chunk

        // ---- scores: S = Q * K^T for this 32-kv chunk (from LDS) -----------
        v8f s0 = {}, s1 = {};
        {
            const _Float16* kr0 = &kbuf[buf][nl][16 * hh];
            const _Float16* kr1 = &kbuf[buf][16 + nl][16 * hh];
            v16h bk00, bk01, bk10, bk11;
#pragma unroll
            for (int e = 0; e < 16; ++e) {
                bk00[e] = kr0[e];  bk01[e] = kr0[32 + e];
                bk10[e] = kr1[e];  bk11[e] = kr1[32 + e];
            }
            s0 = __builtin_amdgcn_wmma_f32_16x16x32_f16(false, aq0, false, bk00, (short)0, s0, false, false);
            s0 = __builtin_amdgcn_wmma_f32_16x16x32_f16(false, aq1, false, bk01, (short)0, s0, false, false);
            s1 = __builtin_amdgcn_wmma_f32_16x16x32_f16(false, aq0, false, bk10, (short)0, s1, false, false);
            s1 = __builtin_amdgcn_wmma_f32_16x16x32_f16(false, aq1, false, bk11, (short)0, s1, false, false);
        }

        // ---- online softmax update -----------------------------------------
        float p0[8], p1[8];
#pragma unroll
        for (int r = 0; r < 8; ++r) {
            const float v0 = s0[r] * sc, v1 = s1[r] * sc;
            float vmax = fmaxf(v0, v1);
#pragma unroll
            for (int msk = 8; msk >= 1; msk >>= 1)      // reduce within 16-lane half
                vmax = fmaxf(vmax, __shfl_xor(vmax, msk, 32));
            const float mn    = fmaxf(mrow[r], vmax);
            const float alpha = __expf(mrow[r] - mn);
            mrow[r] = mn;
            const float e0 = __expf(v0 - mn);
            const float e1 = __expf(v1 - mn);
            float psum = e0 + e1;
#pragma unroll
            for (int msk = 8; msk >= 1; msk >>= 1)
                psum += __shfl_xor(psum, msk, 32);
            lrow[r] = lrow[r] * alpha + psum;
            o[0][r] *= alpha; o[1][r] *= alpha; o[2][r] *= alpha; o[3][r] *= alpha;
            p0[r] = e0; p1[r] = e1;
        }

        // ---- re-stripe P (C-layout -> A-layout) through per-wave LDS -------
#pragma unroll
        for (int r = 0; r < 8; ++r) {
            pbuf[w][8 * hh + r][nl]      = (_Float16)p0[r];
            pbuf[w][8 * hh + r][16 + nl] = (_Float16)p1[r];
        }
        asm volatile("s_wait_dscnt 0" ::: "memory");    // intra-wave: wait, no barrier
        v16h pa;
#pragma unroll
        for (int i = 0; i < 8; ++i)
#pragma unroll
            for (int j = 0; j < 2; ++j)
                pa[2 * i + j] = pbuf[w][nl][a_klocal(i, j, hh)];

        // ---- O += P * V (K-dim = 32 kv, V from LDS) ------------------------
#pragma unroll
        for (int g = 0; g < 4; ++g) {
            const _Float16* vr = &vbuf[buf][16 * g + nl][16 * hh];
            v16h bv;
#pragma unroll
            for (int e = 0; e < 16; ++e) bv[e] = vr[e];
            o[g] = __builtin_amdgcn_wmma_f32_16x16x32_f16(false, pa, false, bv, (short)0, o[g], false, false);
        }

        wait_stage();          // next chunk's async loads landed
        __syncthreads();       // all waves done reading this buf
    }

    // ---- normalize and write out [B,S,D] -----------------------------------
#pragma unroll
    for (int r = 0; r < 8; ++r) {
        const float inv = 1.0f / lrow[r];
        const size_t row = (size_t)b * cS + (m0 + 8 * hh + r);
#pragma unroll
        for (int g = 0; g < 4; ++g)
            out[row * cD + h * cDH + 16 * g + nl] = o[g][r] * inv;
    }
}

// ---------------------------------------------------------------------------
extern "C" void kernel_launch(void* const* d_in, const int* in_sizes, int n_in,
                              void* d_out, int out_size, void* d_ws, size_t ws_size,
                              hipStream_t stream) {
    const float* seq = (const float*)d_in[0];
    const float* qw  = (const float*)d_in[1];
    const float* qb  = (const float*)d_in[2];
    const float* kw  = (const float*)d_in[3];
    const float* kb  = (const float*)d_in[4];
    const float* vw  = (const float*)d_in[5];
    const float* vb  = (const float*)d_in[6];
    float* out = (float*)d_out;

    const size_t per = (size_t)cB * cH * cS * cDH;      // elements per tensor
    _Float16* Q  = (_Float16*)d_ws;
    _Float16* K  = Q + per;
    _Float16* Vt = K + per;

    qkv_proj_kernel<<<cB * cH * cNT, 32, 0, stream>>>(seq, qw, qb, kw, kb, vw, vb, Q, K, Vt);
    flash_attn_kernel<<<cB * cH * (cS / 64), 128, 0, stream>>>(Q, K, Vt, out);
}
